// CBMPredictor_29248727285940
// MI455X (gfx1250) — matrix operand order repaired and checked by reading, not compiled, not run on hardware
//
#include <hip/hip_runtime.h>
#include <hip/hip_bf16.h>

typedef unsigned short u16t;
typedef unsigned int   u32t;

typedef __attribute__((ext_vector_type(16))) __bf16 v16bf;
typedef __attribute__((ext_vector_type(8)))  float  v8f;

#define N_NODES   100000
#define N_EDGES   1600000
#define N_GRAPHS  256
#define HDIM      128
#define KTILES    9          // 288 padded K / 32
#define EPSV      1e-5f

union Frag { v16bf v; uint4 q[2]; u16t u[16]; };

__device__ __forceinline__ u16t f2bf(float f) {
  union { float f; u32t u; } x; x.f = f;
  u32t r = x.u + 0x7FFFu + ((x.u >> 16) & 1u);   // round-to-nearest-even
  return (u16t)(r >> 16);
}
__device__ __forceinline__ float sigm(float x) { return 1.f / (1.f + __expf(-x)); }
__device__ __forceinline__ float siluf(float x) { return x * sigm(x); }
__device__ __forceinline__ float softplusf(float x) {
  if (x > 20.f) return x;
  if (x < -20.f) return __expf(x);
  return log1pf(__expf(x));
}
__device__ __forceinline__ float cleanf(float v) {  // nan->0, clamp +-inf
  if (v != v) return 0.f;
  return fminf(fmaxf(v, -3.0e38f), 3.0e38f);
}

// ---------------------------------------------------------------- encoder
__global__ void encode_kernel(const float* __restrict__ x,
                              const float* __restrict__ Win,
                              const float* __restrict__ b_in,
                              float* __restrict__ h, u16t* __restrict__ hb) {
  int n = blockIdx.x, t = threadIdx.x;
  float acc = b_in[t];
#pragma unroll
  for (int d = 0; d < 7; ++d) {
    float xv = x[n * 7 + d];
    xv = (xv != xv) ? 0.f : (xv > 1e30f ? 3.f : (xv < -1e30f ? -3.f : xv));
    acc += xv * Win[d * HDIM + t];
  }
  float hv = siluf(acc);
  size_t idx = (size_t)n * HDIM + t;
  h[idx]  = hv;
  hb[idx] = f2bf(hv);
}

__global__ void zero_kernel(float* __restrict__ p, int n) {
  int i = blockIdx.x * blockDim.x + threadIdx.x;
  if (i < n) p[i] = 0.f;
}

// ------------------------------------------ weight swizzle to WMMA fragments
// tile = (mat*KTILES + kt)*8 + nt ; lane fragment = 16 contiguous bf16
__global__ void convw_kernel(const float* __restrict__ Wf_l,
                             const float* __restrict__ Ws_l,
                             u16t* __restrict__ wfrag) {
  int tile = blockIdx.x;                 // 0..143
  int nt   = tile & 7;
  int kt   = (tile >> 3) % KTILES;
  int mat  = tile / (8 * KTILES);
  int lane = threadIdx.x >> 4;           // 0..31
  int j    = threadIdx.x & 15;           // 0..15
  int k = kt * 32 + (lane >> 4) * 8 + (j & 7) + (j >> 3) * 16;  // B layout: khalf*8 + j%8 + (j/8)*16
  int n = nt * 16 + (lane & 15);
  const float* W = mat ? Ws_l : Wf_l;
  float v = (k < 260) ? W[(size_t)k * HDIM + n] : 0.f;
  wfrag[(size_t)tile * 512 + lane * 16 + j] = f2bf(v);
}

// ---------------------------------------------------------- WMMA edge GEMM
// block = 128 threads (4 waves); wave handles 16 edges x 128 outputs for BOTH
// weight matrices. A = [h_bf[dst](128) | h_bf[src](128) | ea(4) | zeros(28)].
// A fragments double-buffered across k-steps (L2-latency gathers); B fragment
// pairs double-buffered across n-tiles so loads overlap the WMMA pipe.
__global__ void __launch_bounds__(128) edge_kernel(
    const u16t* __restrict__ hb, const float* __restrict__ eattr,
    const int* __restrict__ eidx, const u16t* __restrict__ wfrag,
    const float* __restrict__ bfv, const float* __restrict__ bsv,
    float* __restrict__ agg) {
  const int lane  = threadIdx.x & 31;
  const int wave  = threadIdx.x >> 5;
  const int khalf = lane >> 4;
  const int m     = lane & 15;
  const int ebase = blockIdx.x * 64 + wave * 16;
  const int e     = ebase + m;
  const int si = eidx[e];             // edge_index[0] = source j
  const int di = eidx[N_EDGES + e];   // edge_index[1] = target i

  const u16t* wl = wfrag + (size_t)lane * 16;

  auto load_a = [&](Frag& f, int kt) {
    // two contiguous 8x-bf16 runs per the 16-bit A 16x32 layout; region
    // boundaries (128/256) are 8-aligned so each run has a single source.
    int o1 = kt * 32 + khalf * 8;
    int o2 = o1 + 16;
    const u16t* p1 = (o1 < 128) ? hb + (size_t)di * HDIM + o1
                                : hb + (size_t)si * HDIM + (o1 - 128);
    const u16t* p2 = (o2 < 128) ? hb + (size_t)di * HDIM + o2
                                : hb + (size_t)si * HDIM + (o2 - 128);
    f.q[0] = *(const uint4*)p1;
    f.q[1] = *(const uint4*)p2;
  };
  auto load_b = [&](Frag& f, int mat, int kt, int nt) {
    const uint4* p =
        (const uint4*)(wl + (size_t)((mat * KTILES + kt) * 8 + nt) * 512);
    f.q[0] = p[0];
    f.q[1] = p[1];
  };

  // K-tail fragment (K=256..287): edge_attr in first 4 lanes of khalf 0
  Frag a8;
  a8.q[0] = make_uint4(0u, 0u, 0u, 0u);
  a8.q[1] = make_uint4(0u, 0u, 0u, 0u);
  if (khalf == 0) {
    float4 ev = *(const float4*)(eattr + (size_t)e * 4);
    float c0 = (ev.x != ev.x) ? 0.f : (ev.x > 1e30f ? 1.f : (ev.x < -1e30f ? 0.f : ev.x));
    float c1 = (ev.y != ev.y) ? 0.f : (ev.y > 1e30f ? 1.f : (ev.y < -1e30f ? 0.f : ev.y));
    float c2 = (ev.z != ev.z) ? 0.f : (ev.z > 1e30f ? 1.f : (ev.z < -1e30f ? 0.f : ev.z));
    float c3 = (ev.w != ev.w) ? 0.f : (ev.w > 1e30f ? 1.f : (ev.w < -1e30f ? 0.f : ev.w));
    a8.u[0] = f2bf(c0); a8.u[1] = f2bf(c1); a8.u[2] = f2bf(c2); a8.u[3] = f2bf(c3);
  }

  v8f accf[8] = {};
  v8f accs[8] = {};

  Frag a[2], bF[2], bS[2];
  load_a(a[0], 0);
  load_b(bF[0], 0, 0, 0);
  load_b(bS[0], 1, 0, 0);

#pragma unroll
  for (int kt = 0; kt < KTILES; ++kt) {
    if (kt + 1 < 8) load_a(a[(kt + 1) & 1], kt + 1);
    __builtin_prefetch((const void*)(wl + (size_t)((kt + 1) * 8) * 512), 0, 0);
#pragma unroll
    for (int nt = 0; nt < 8; ++nt) {
      const int nk = (nt == 7) ? kt + 1 : kt;
      const int nn = (nt == 7) ? 0 : nt + 1;
      if (nk < KTILES) {
        load_b(bF[(nt + 1) & 1], 0, nk, nn);
        load_b(bS[(nt + 1) & 1], 1, nk, nn);
      }
      const Frag& ac = (kt == 8) ? a8 : a[kt & 1];
      accf[nt] = __builtin_amdgcn_wmma_f32_16x16x32_bf16(
          false, ac.v, false, bF[nt & 1].v, (short)0, accf[nt], false, false);
      accs[nt] = __builtin_amdgcn_wmma_f32_16x16x32_bf16(
          false, ac.v, false, bS[nt & 1].v, (short)0, accs[nt], false, false);
    }
  }

  // epilogue: msg = sigmoid(f)*softplus(s); scatter-add by dst (segment_sum)
  int drow[8];
#pragma unroll
  for (int r = 0; r < 8; ++r)
    drow[r] = eidx[N_EDGES + ebase + khalf * 8 + r];
#pragma unroll
  for (int nt = 0; nt < 8; ++nt) {
    int n = nt * 16 + m;          // C/D layout: N = lane%16, M = khalf*8 + r
    float bfn = bfv[n], bsn = bsv[n];
#pragma unroll
    for (int r = 0; r < 8; ++r) {
      float msg = sigm(accf[nt][r] + bfn) * softplusf(accs[nt][r] + bsn);
      atomicAdd(agg + (size_t)drow[r] * HDIM + n, msg);
    }
  }
}

// ----------------------------------------------------- residual + LayerNorm
__global__ void node_update_kernel(float* __restrict__ h, u16t* __restrict__ hb,
                                   float* __restrict__ agg,
                                   const float* __restrict__ g,
                                   const float* __restrict__ b) {
  __shared__ float red[8];
  int n = blockIdx.x, t = threadIdx.x;
  size_t idx = (size_t)n * HDIM + t;
  float hv = h[idx];
  float av = agg[idx];
  agg[idx] = 0.f;                       // reset for next layer
  float s = siluf(hv + av) + hv;

  float v = s;
  for (int i = 16; i > 0; i >>= 1) v += __shfl_xor(v, i, 32);
  if ((t & 31) == 0) red[t >> 5] = v;
  __syncthreads();
  float mean = (red[0] + red[1] + red[2] + red[3]) * (1.f / HDIM);
  __syncthreads();
  float d = s - mean;
  float v2 = d * d;
  for (int i = 16; i > 0; i >>= 1) v2 += __shfl_xor(v2, i, 32);
  if ((t & 31) == 0) red[t >> 5] = v2;
  __syncthreads();
  float var = (red[0] + red[1] + red[2] + red[3]) * (1.f / HDIM);
  float out = d * rsqrtf(var + EPSV) * g[t] + b[t];
  h[idx]  = out;
  hb[idx] = f2bf(out);
}

// ---------------------------------------------------------------- pooling
__global__ void pool_kernel(const float* __restrict__ h,
                            const int* __restrict__ bid,
                            float* __restrict__ ps, float* __restrict__ cnt) {
  int n = blockIdx.x, t = threadIdx.x;
  int gr = bid[n];
  atomicAdd(ps + (size_t)gr * HDIM + t, h[(size_t)n * HDIM + t]);
  if (t == 0) atomicAdd(cnt + gr, 1.f);
}

// --------------------------------------------------------------- head
__device__ __forceinline__ float block_sum6(float v, float* red) {
  for (int i = 16; i > 0; i >>= 1) v += __shfl_xor(v, i, 32);
  int w = threadIdx.x >> 5;
  if ((threadIdx.x & 31) == 0) red[w] = v;
  __syncthreads();
  if (threadIdx.x == 0) {
    float s = 0.f;
    for (int i = 0; i < 6; ++i) s += red[i];
    red[7] = s;
  }
  __syncthreads();
  float r = red[7];
  __syncthreads();
  return r;
}

__global__ void __launch_bounds__(192) head_kernel(
    const float* __restrict__ ps, const float* __restrict__ cnt,
    const float* __restrict__ tda,
    const float* __restrict__ Wout, const float* __restrict__ bout,
    const float* __restrict__ Wt1, const float* __restrict__ bt1,
    const float* __restrict__ tg, const float* __restrict__ tb,
    const float* __restrict__ Wt2, const float* __restrict__ bt2,
    const float* __restrict__ fng, const float* __restrict__ fnb,
    const float* __restrict__ Wg1, const float* __restrict__ bg1,
    const float* __restrict__ Wg2, const float* __restrict__ bg2,
    const float* __restrict__ Wpl, const float* __restrict__ bpl,
    const float* __restrict__ Wq1, const float* __restrict__ bq1,
    const float* __restrict__ Wq2, const float* __restrict__ bq2,
    float* __restrict__ out) {
  __shared__ float sa[192], sz[192], st[32], red[8], sg[64];
  int g = blockIdx.x, t = threadIdx.x;

  if (t < 128) sa[t] = ps[(size_t)g * HDIM + t] / fmaxf(cnt[g], 1.f);
  if (t < 32) {
    float v = tda[g * 32 + t];
    st[t] = (v != v) ? 0.f : (v > 1e30f ? 3.f : (v < -1e30f ? -3.f : v));
  }
  __syncthreads();

  float zg = 0.f;
  if (t < 128) {
    float acc = bout[t];
    for (int k = 0; k < 128; ++k) acc += sa[k] * Wout[k * HDIM + t];
    zg = cleanf(acc);
  }
  __syncthreads();
  if (t < 128) sz[t] = zg;

  // TDA projector: 32 -> 128 silu -> LN -> 64
  float u = 0.f;
  if (t < 128) {
    float acc = bt1[t];
    for (int k = 0; k < 32; ++k) acc += st[k] * Wt1[k * HDIM + t];
    u = siluf(acc);
  }
  float mean = block_sum6((t < 128) ? u : 0.f, red) * (1.f / 128.f);
  float dd = (t < 128) ? (u - mean) : 0.f;
  float var = block_sum6(dd * dd, red) * (1.f / 128.f);
  if (t < 128) sa[t] = dd * rsqrtf(var + EPSV) * tg[t] + tb[t];
  __syncthreads();
  if (t < 64) {
    float acc = bt2[t];
    for (int k = 0; k < 128; ++k) acc += sa[k] * Wt2[k * 64 + t];
    sz[128 + t] = cleanf(acc);
  }
  __syncthreads();

  // fusion LayerNorm over 192
  float zz = sz[t];
  float m2 = block_sum6(zz, red) * (1.f / 192.f);
  float d2 = zz - m2;
  float v2 = block_sum6(d2 * d2, red) * (1.f / 192.f);
  float zf = d2 * rsqrtf(v2 + EPSV) * fng[t] + fnb[t];
  __syncthreads();
  sz[t] = zf;
  out[256 + (size_t)g * 192 + t] = zf;
  __syncthreads();

  // gates + linear term
  if (t < 16) {
    float acc = bg1[t];
    for (int k = 0; k < 192; ++k) acc += sz[k] * Wg1[k * 16 + t];
    sg[t] = siluf(acc);
  }
  if (t < 4) {
    float acc = bpl[t];
    for (int k = 0; k < 192; ++k) acc += sz[k] * Wpl[t * 192 + k];
    sg[32 + t] = acc;                         // lin
  }
  __syncthreads();
  if (t < 4) {
    float acc = bg2[t];
    for (int k = 0; k < 16; ++k) acc += sg[k] * Wg2[k * 4 + t];
    sg[40 + t] = acc;                         // logits
  }
  __syncthreads();
  if (t == 0) {
    float mx = fmaxf(fmaxf(sg[40], sg[41]), fmaxf(sg[42], sg[43]));
    float s = 0.f;
    for (int k = 0; k < 4; ++k) { float ee = __expf(sg[40 + k] - mx); sg[48 + k] = ee; s += ee; }
    for (int k = 0; k < 4; ++k) sg[48 + k] /= s;   // gates
  }
  __syncthreads();

  // quadratic term
  float quad[4];
  for (int k = 0; k < 4; ++k) {
    float part = 0.f;
    if (t < 96) {
      float acc = bq1[k * 96 + t];
      for (int d = 0; d < 192; ++d) acc += sz[d] * Wq1[((size_t)k * 192 + d) * 96 + t];
      part = siluf(acc) * Wq2[k * 96 + t];
    }
    quad[k] = block_sum6(part, red) + bq2[k];
  }
  if (t == 0) {
    float y = 0.f;
    for (int k = 0; k < 4; ++k) y += sg[48 + k] * (sg[32 + k] + quad[k]);
    out[g] = y;
  }
}

// ---------------------------------------------------------------- launch
extern "C" void kernel_launch(void* const* d_in, const int* in_sizes, int n_in,
                              void* d_out, int out_size, void* d_ws, size_t ws_size,
                              hipStream_t stream) {
  const float* x    = (const float*)d_in[0];
  const float* ea   = (const float*)d_in[1];
  const float* tda  = (const float*)d_in[2];
  const int*   eidx = (const int*)d_in[3];
  const int*   bid  = (const int*)d_in[4];
  const float* Win  = (const float*)d_in[5];
  const float* b_in = (const float*)d_in[6];
  const float* Wf   = (const float*)d_in[7];
  const float* bfv  = (const float*)d_in[8];
  const float* Ws   = (const float*)d_in[9];
  const float* bsv  = (const float*)d_in[10];
  const float* lng  = (const float*)d_in[11];
  const float* lnb  = (const float*)d_in[12];
  const float* Wout = (const float*)d_in[13];
  const float* bout = (const float*)d_in[14];
  const float* Wt1  = (const float*)d_in[15];
  const float* bt1  = (const float*)d_in[16];
  const float* tg   = (const float*)d_in[17];
  const float* tb   = (const float*)d_in[18];
  const float* Wt2  = (const float*)d_in[19];
  const float* bt2  = (const float*)d_in[20];
  const float* fng  = (const float*)d_in[21];
  const float* fnb  = (const float*)d_in[22];
  const float* Wg1  = (const float*)d_in[23];
  const float* bg1  = (const float*)d_in[24];
  const float* Wg2  = (const float*)d_in[25];
  const float* bg2  = (const float*)d_in[26];
  const float* Wpl  = (const float*)d_in[27];
  const float* bpl  = (const float*)d_in[28];
  const float* Wq1  = (const float*)d_in[29];
  const float* bq1  = (const float*)d_in[30];
  const float* Wq2  = (const float*)d_in[31];
  const float* bq2  = (const float*)d_in[32];
  float* out = (float*)d_out;

  // workspace layout (~128.3 MB)
  float* h      = (float*)d_ws;
  float* agg    = h + (size_t)N_NODES * HDIM;
  u16t*  hb     = (u16t*)(agg + (size_t)N_NODES * HDIM);
  u16t*  wfrag  = hb + (size_t)N_NODES * HDIM;             // 2*9*8*512 u16
  float* pooled = (float*)(wfrag + 2 * KTILES * 8 * 512);
  float* cntp   = pooled + N_GRAPHS * HDIM;

  encode_kernel<<<N_NODES, 128, 0, stream>>>(x, Win, b_in, h, hb);
  zero_kernel<<<(N_NODES * HDIM) / 256, 256, 0, stream>>>(agg, N_NODES * HDIM);

  for (int l = 0; l < 4; ++l) {
    convw_kernel<<<2 * KTILES * 8, 512, 0, stream>>>(
        Wf + (size_t)l * 260 * HDIM, Ws + (size_t)l * 260 * HDIM, wfrag);
    edge_kernel<<<N_EDGES / 64, 128, 0, stream>>>(
        hb, ea, eidx, wfrag, bfv + l * HDIM, bsv + l * HDIM, agg);
    node_update_kernel<<<N_NODES, 128, 0, stream>>>(
        h, hb, agg, lng + l * HDIM, lnb + l * HDIM);
  }

  zero_kernel<<<(N_GRAPHS * HDIM + N_GRAPHS + 255) / 256, 256, 0, stream>>>(
      pooled, N_GRAPHS * HDIM + N_GRAPHS);
  pool_kernel<<<N_NODES, 128, 0, stream>>>(h, bid, pooled, cntp);

  head_kernel<<<N_GRAPHS, 192, 0, stream>>>(
      pooled, cntp, tda, Wout, bout, Wt1, bt1, tg, tb, Wt2, bt2,
      fng, fnb, Wg1, bg1, Wg2, bg2, Wpl, bpl, Wq1, bq1, Wq2, bq2, out);
}
